// SegDecoder_70660801954537
// MI455X (gfx1250) — matrix operand order repaired and checked by reading, not compiled, not run on hardware
//
#include <hip/hip_runtime.h>

typedef __attribute__((ext_vector_type(16))) __bf16 v16bf;
typedef __attribute__((ext_vector_type(8)))  float  v8f;
typedef __attribute__((ext_vector_type(4)))  unsigned int v4u;
typedef unsigned short u16;
typedef unsigned int   u32;

union Frag { v16bf bf; v4u u[2]; };

__device__ __forceinline__ u16 f2bf(float f) {
  u32 u = __builtin_bit_cast(u32, f);
  u += 0x7FFFu + ((u >> 16) & 1u);
  return (u16)(u >> 16);
}
__device__ __forceinline__ float bf2f(u16 h) {
  u32 u = ((u32)h) << 16;
  return __builtin_bit_cast(float, u);
}
__device__ __forceinline__ v8f vzero() {
  v8f v;
#pragma unroll
  for (int i = 0; i < 8; ++i) v[i] = 0.f;
  return v;
}

#define WMMA_BF16(a, b, c) \
  __builtin_amdgcn_wmma_f32_16x16x32_bf16(false, (a), false, (b), (short)0, (c), false, false)

// ---------------- prep kernels ----------------

__global__ void k_f32_to_bf16(const float* __restrict__ src, u16* __restrict__ dst, long n) {
  long i = (long)blockIdx.x * 256 + threadIdx.x;
  if (i < n) dst[i] = f2bf(src[i]);
}

// W[K,N] f32 row-major -> WT[N,K] bf16 row-major
__global__ void k_transpose_w(const float* __restrict__ W, u16* __restrict__ WT, int K, int N) {
  long idx = (long)blockIdx.x * 256 + threadIdx.x;
  if (idx >= (long)K * N) return;
  int n = (int)(idx / K);
  int k = (int)(idx % K);
  WT[idx] = f2bf(W[(size_t)k * N + n]);
}

__global__ void k_init_x(const float* __restrict__ q, float* __restrict__ x, u16* __restrict__ xb, long n) {
  long i = (long)blockIdx.x * 256 + threadIdx.x;
  if (i < n) { float v = q[i]; x[i] = v; xb[i] = f2bf(v); }
}

// ---------------- generic WMMA GEMM ----------------
// C[M,N] = A[M,K] (bf16) * BT[N,K]^T (bf16) + bias[N]
// mode 0: f32 out; 1: bf16 out; 2: bf16 out transposed per-batch (V layout Vt[b][c][n], N==512, 4096 rows/b);
// mode 3: bf16 out with ReLU.
__global__ __launch_bounds__(32) void k_gemm(const u16* __restrict__ A, const u16* __restrict__ BT,
                                             const float* __restrict__ bias, int N, int K, int mode,
                                             float* __restrict__ Cf, u16* __restrict__ Cb) {
  const int lane = threadIdx.x;
  const int sub = lane >> 4;    // which 16-lane half
  const int ln  = lane & 15;
  const size_t m0 = (size_t)blockIdx.x * 32;
  const int    n0 = blockIdx.y * 64;

  v8f acc[2][4];
#pragma unroll
  for (int i = 0; i < 2; ++i)
#pragma unroll
    for (int j = 0; j < 4; ++j) acc[i][j] = vzero();

  for (int kk = 0; kk < K; kk += 32) {
    Frag a[2], b[4];
#pragma unroll
    for (int i = 0; i < 2; ++i) {
      const u16* ap = A + (m0 + i * 16 + ln) * (size_t)K + kk + (sub ? 8 : 0);
      a[i].u[0] = *(const v4u*)ap;
      a[i].u[1] = *(const v4u*)(ap + 16);
    }
#pragma unroll
    for (int j = 0; j < 4; ++j) {
      const u16* bp = BT + (size_t)(n0 + j * 16 + ln) * K + kk + (sub ? 8 : 0);
      b[j].u[0] = *(const v4u*)bp;
      b[j].u[1] = *(const v4u*)(bp + 16);
    }
#pragma unroll
    for (int i = 0; i < 2; ++i)
#pragma unroll
      for (int j = 0; j < 4; ++j) acc[i][j] = WMMA_BF16(a[i].bf, b[j].bf, acc[i][j]);
  }

#pragma unroll
  for (int i = 0; i < 2; ++i)
#pragma unroll
    for (int j = 0; j < 4; ++j)
#pragma unroll
      for (int r = 0; r < 8; ++r) {
        size_t m = m0 + i * 16 + r + sub * 8;
        int n = n0 + j * 16 + ln;
        float v = acc[i][j][r] + bias[n];
        if (mode == 3) v = fmaxf(v, 0.f);
        if (mode == 0) Cf[m * N + n] = v;
        else if (mode == 2) {
          size_t bb = m >> 12, nn = m & 4095;
          Cb[(bb * 512 + (size_t)n) * 4096 + nn] = f2bf(v);
        } else Cb[m * N + n] = f2bf(v);
      }
}

// ---------------- affinity gate ----------------
// aff[b,n] = sigmoid((SCALE/H) * dot(extra[b,:], Kp[b,n,:]))
__global__ void k_aff(const u16* __restrict__ Kp, const float* __restrict__ extra,
                      float* __restrict__ affws, float* __restrict__ affout) {
  int t = blockIdx.x * 256 + threadIdx.x;     // b*4096 + n, 65536 total
  int b = t >> 12;
  const u16* kp = Kp + (size_t)t * 512;
  const float* e = extra + (size_t)b * 512;
  float s = 0.f;
#pragma unroll 4
  for (int c = 0; c < 512; ++c) s += bf2f(kp[c]) * e[c];
  float z = s * (0.125f / 8.0f);              // SCALE / H
  float a = 1.f / (1.f + __expf(-z));
  affws[t] = a;
  affout[t] = a;
}

// ---------------- attention (per layer) ----------------
// grid (16 kslices, 4 qtiles, 16 batch), 1 wave per block, all 8 heads looped inside.
__global__ __launch_bounds__(32) void k_attn(const u16* __restrict__ Qp, const u16* __restrict__ Kp,
                                             const u16* __restrict__ Vt, const float* __restrict__ aff,
                                             float* attn_out, float* __restrict__ Opart,
                                             float* __restrict__ RSpart) {
  __shared__ __align__(16) u16 plds[16 * 32];
  const int ks = blockIdx.x, qt = blockIdx.y, b = blockIdx.z;
  const int lane = threadIdx.x, sub = lane >> 4, ln = lane & 15;

  for (int h = 0; h < 8; ++h) {
    Frag aq[2];
    const u16* qpr = Qp + (size_t)(b * 64 + qt * 16 + ln) * 512 + h * 64;
#pragma unroll
    for (int s2 = 0; s2 < 2; ++s2) {
      const u16* ap = qpr + s2 * 32 + (sub ? 8 : 0);
      aq[s2].u[0] = *(const v4u*)ap;
      aq[s2].u[1] = *(const v4u*)(ap + 16);
    }
    v8f oacc[4];
#pragma unroll
    for (int j = 0; j < 4; ++j) oacc[j] = vzero();
    float rs[8] = {0.f, 0.f, 0.f, 0.f, 0.f, 0.f, 0.f, 0.f};

    for (int dk = 0; dk < 8; ++dk) {
      int k0 = ks * 256 + dk * 32;
#pragma unroll
      for (int t = 0; t < 2; ++t) {
        int kt = k0 + t * 16;
        v8f s = vzero();
        const u16* kpr = Kp + (size_t)(b * 4096 + kt + ln) * 512 + h * 64;
#pragma unroll
        for (int s2 = 0; s2 < 2; ++s2) {
          Frag bk;
          const u16* bp = kpr + s2 * 32 + (sub ? 8 : 0);
          bk.u[0] = *(const v4u*)bp;
          bk.u[1] = *(const v4u*)(bp + 16);
          s = WMMA_BF16(aq[s2].bf, bk.bf, s);
        }
        float av = aff[b * 4096 + kt + ln];
        float* attp = attn_out + (size_t)(b * 64 + qt * 16) * 4096 + kt + ln;
#pragma unroll
        for (int r = 0; r < 8; ++r) {
          float sv = s[r] * 0.125f;           // SCALE
          int row = r + sub * 8;
          size_t o = (size_t)row * 4096;
          float acca = sv * 0.125f;           // S / H
          if (h) acca += attp[o];
          attp[o] = acca;
          float ev = __expf(sv);              // |S| small (0.02-scaled weights): no max needed
          rs[r] += ev;
          plds[row * 32 + t * 16 + ln] = f2bf(ev * av);
        }
      }
      __syncthreads();
      Frag pa;
      const u16* pp = &plds[ln * 32 + (sub ? 8 : 0)];
      pa.u[0] = *(const v4u*)pp;
      pa.u[1] = *(const v4u*)(pp + 16);
      __syncthreads();
#pragma unroll
      for (int j = 0; j < 4; ++j) {
        const u16* vp = Vt + (size_t)(b * 512 + h * 64 + j * 16 + ln) * 4096 + k0 + (sub ? 8 : 0);
        Frag bv;
        bv.u[0] = *(const v4u*)vp;
        bv.u[1] = *(const v4u*)(vp + 16);
        oacc[j] = WMMA_BF16(pa.bf, bv.bf, oacc[j]);
      }
    }
    // reduce softmax denominator across the 16 lanes of each half-wave
#pragma unroll
    for (int r = 0; r < 8; ++r) {
      rs[r] += __shfl_xor(rs[r], 1, 16);
      rs[r] += __shfl_xor(rs[r], 2, 16);
      rs[r] += __shfl_xor(rs[r], 4, 16);
      rs[r] += __shfl_xor(rs[r], 8, 16);
    }
    size_t pidx = (((size_t)b * 8 + h) * 4 + qt) * 16 + ks;
    float* op = Opart + pidx * 1024;
#pragma unroll
    for (int j = 0; j < 4; ++j)
#pragma unroll
      for (int r = 0; r < 8; ++r) op[(r + sub * 8) * 64 + j * 16 + ln] = oacc[j][r];
    if (ln == 0) {
      float* rp = RSpart + pidx * 16;
#pragma unroll
      for (int r = 0; r < 8; ++r) rp[r + sub * 8] = rs[r];
    }
  }
}

// reduce over 16 key-slices, divide by softmax denom, emit bf16 o for Wo GEMM
__global__ __launch_bounds__(64) void k_attn_reduce(const float* __restrict__ Opart,
                                                    const float* __restrict__ RSpart,
                                                    u16* __restrict__ obf) {
  int idx = blockIdx.x;          // (b*8 + h)*4 + qt
  int d = threadIdx.x;
  int qt = idx & 3, h = (idx >> 2) & 7, b = idx >> 5;
  for (int row = 0; row < 16; ++row) {
    float z = 0.f, num = 0.f;
    for (int ks = 0; ks < 16; ++ks) {
      size_t p = (size_t)idx * 16 + ks;
      z += RSpart[p * 16 + row];
      num += Opart[p * 1024 + (size_t)row * 64 + d];
    }
    obf[(size_t)(b * 64 + qt * 16 + row) * 512 + h * 64 + d] = f2bf(num / z);
  }
}

// ---------------- residual + LayerNorm ----------------
__global__ __launch_bounds__(128) void k_ln(const float* xin, const float* __restrict__ add,
                                            const float* __restrict__ g, const float* __restrict__ be,
                                            float* xout, u16* __restrict__ xb, float* outs) {
  __shared__ float s1[128], s2[128];
  int row = blockIdx.x, tid = threadIdx.x;
  const float* xi = xin + (size_t)row * 512;
  const float* ad = add + (size_t)row * 512;
  float v[4], a1 = 0.f, a2 = 0.f;
#pragma unroll
  for (int j = 0; j < 4; ++j) {
    int c = tid + 128 * j;
    v[j] = xi[c] + ad[c];
    a1 += v[j];
    a2 += v[j] * v[j];
  }
  s1[tid] = a1; s2[tid] = a2;
  __syncthreads();
  for (int st = 64; st > 0; st >>= 1) {
    if (tid < st) { s1[tid] += s1[tid + st]; s2[tid] += s2[tid + st]; }
    __syncthreads();
  }
  float mu = s1[0] * (1.f / 512.f);
  float var = s2[0] * (1.f / 512.f) - mu * mu;
  float rstd = rsqrtf(var + 1e-5f);
#pragma unroll
  for (int j = 0; j < 4; ++j) {
    int c = tid + 128 * j;
    float y = g[c] * (v[j] - mu) * rstd + be[c];
    xout[(size_t)row * 512 + c] = y;
    xb[(size_t)row * 512 + c] = f2bf(y);
    if (outs) outs[(size_t)row * 512 + c] = y;
  }
}

// ---------------- launch ----------------

extern "C" void kernel_launch(void* const* d_in, const int* in_sizes, int n_in,
                              void* d_out, int out_size, void* d_ws, size_t ws_size,
                              hipStream_t stream) {
  (void)in_sizes; (void)n_in; (void)out_size; (void)ws_size;
  const float* queries = (const float*)d_in[0];
  const float* feat    = (const float*)d_in[1];
  const float* extra   = (const float*)d_in[2];
  const float* Wq = (const float*)d_in[3];  const float* bq = (const float*)d_in[4];
  const float* Wk = (const float*)d_in[5];  const float* bk = (const float*)d_in[6];
  const float* Wv = (const float*)d_in[7];  const float* bv = (const float*)d_in[8];
  const float* Wo = (const float*)d_in[9];  const float* bo = (const float*)d_in[10];
  const float* W1 = (const float*)d_in[11]; const float* b1 = (const float*)d_in[12];
  const float* W2 = (const float*)d_in[13]; const float* b2 = (const float*)d_in[14];
  const float* g2 = (const float*)d_in[15]; const float* be2 = (const float*)d_in[16];
  const float* g3 = (const float*)d_in[17]; const float* be3 = (const float*)d_in[18];

  char* ws = (char*)d_ws;
  size_t off = 0;
  auto carve = [&](size_t bytes) -> char* {
    char* p = ws + off;
    off += (bytes + 255) & ~(size_t)255;
    return p;
  };

  u16* WT[3][6];
  for (int l = 0; l < 3; ++l) {
    WT[l][0] = (u16*)carve(512 * 512 * 2);      // WqT [512,512]
    WT[l][1] = (u16*)carve(512 * 512 * 2);      // WkT
    WT[l][2] = (u16*)carve(512 * 512 * 2);      // WvT
    WT[l][3] = (u16*)carve(512 * 512 * 2);      // WoT
    WT[l][4] = (u16*)carve((size_t)2048 * 512 * 2);  // W1T [2048,512]
    WT[l][5] = (u16*)carve((size_t)512 * 2048 * 2);  // W2T [512,2048]
  }
  u16* featb = (u16*)carve((size_t)16 * 4096 * 512 * 2);
  u16* Kp    = (u16*)carve((size_t)16 * 4096 * 512 * 2);
  u16* Vt    = (u16*)carve((size_t)16 * 512 * 4096 * 2);
  u16* Qp    = (u16*)carve((size_t)1024 * 512 * 2);
  u16* xb    = (u16*)carve((size_t)1024 * 512 * 2);
  u16* obf   = (u16*)carve((size_t)1024 * 512 * 2);
  u16* h1    = (u16*)carve((size_t)1024 * 2048 * 2);
  float* x   = (float*)carve((size_t)1024 * 512 * 4);
  float* tmp = (float*)carve((size_t)1024 * 512 * 4);
  float* affws  = (float*)carve((size_t)65536 * 4);
  float* Opart  = (float*)carve((size_t)8192 * 1024 * 4);
  float* RSpart = (float*)carve((size_t)8192 * 16 * 4);

  float* outs  = (float*)d_out;                                   // [3,16,64,512]
  float* attns = outs + (size_t)3 * 1024 * 512;                   // [3,16,64,4096]
  float* affs  = attns + (size_t)3 * 16 * 64 * 4096;              // [3,16,1,1,4096]

  // ---- prep: weight transposes + input conversions ----
  for (int l = 0; l < 3; ++l) {
    k_transpose_w<<<(512 * 512) / 256, 256, 0, stream>>>(Wq + (size_t)l * 262144, WT[l][0], 512, 512);
    k_transpose_w<<<(512 * 512) / 256, 256, 0, stream>>>(Wk + (size_t)l * 262144, WT[l][1], 512, 512);
    k_transpose_w<<<(512 * 512) / 256, 256, 0, stream>>>(Wv + (size_t)l * 262144, WT[l][2], 512, 512);
    k_transpose_w<<<(512 * 512) / 256, 256, 0, stream>>>(Wo + (size_t)l * 262144, WT[l][3], 512, 512);
    k_transpose_w<<<(512 * 2048) / 256, 256, 0, stream>>>(W1 + (size_t)l * 1048576, WT[l][4], 512, 2048);
    k_transpose_w<<<(512 * 2048) / 256, 256, 0, stream>>>(W2 + (size_t)l * 1048576, WT[l][5], 2048, 512);
  }
  k_f32_to_bf16<<<(16 * 4096 * 512) / 256, 256, 0, stream>>>(feat, featb, (long)16 * 4096 * 512);
  k_init_x<<<(1024 * 512) / 256, 256, 0, stream>>>(queries, x, xb, (long)1024 * 512);

  for (int l = 0; l < 3; ++l) {
    // K, V projections (dominant GEMMs): M=65536, N=512, K=512
    k_gemm<<<dim3(2048, 8), 32, 0, stream>>>(featb, WT[l][1], bk + l * 512, 512, 512, 1, nullptr, Kp);
    k_gemm<<<dim3(2048, 8), 32, 0, stream>>>(featb, WT[l][2], bv + l * 512, 512, 512, 2, nullptr, Vt);
    // affinity gate (collapses to a C-dim dot with extra_token)
    k_aff<<<256, 256, 0, stream>>>(Kp, extra, affws, affs + (size_t)l * 65536);
    // Q projection: M=1024
    k_gemm<<<dim3(32, 8), 32, 0, stream>>>(xb, WT[l][0], bq + l * 512, 512, 512, 1, nullptr, Qp);
    // attention: scores + attns output + gated softmax-weighted V partials
    k_attn<<<dim3(16, 4, 16), 32, 0, stream>>>(Qp, Kp, Vt, affws,
                                               attns + (size_t)l * 16 * 64 * 4096, Opart, RSpart);
    k_attn_reduce<<<512, 64, 0, stream>>>(Opart, RSpart, obf);
    // output projection, residual + LN
    k_gemm<<<dim3(32, 8), 32, 0, stream>>>(obf, WT[l][3], bo + l * 512, 512, 512, 0, tmp, nullptr);
    k_ln<<<1024, 128, 0, stream>>>(x, tmp, g2 + l * 512, be2 + l * 512, x, xb, nullptr);
    // FFN: relu(x@W1+b1)@W2+b2, residual + LN (also emits outs[l])
    k_gemm<<<dim3(32, 32), 32, 0, stream>>>(xb, WT[l][4], b1 + (size_t)l * 2048, 2048, 512, 3, nullptr, h1);
    k_gemm<<<dim3(32, 8), 32, 0, stream>>>(h1, WT[l][5], b2 + l * 512, 512, 2048, 0, tmp, nullptr);
    k_ln<<<1024, 128, 0, stream>>>(x, tmp, g3 + l * 512, be3 + l * 512, x, xb,
                                   outs + (size_t)l * 1024 * 512);
  }
}
